// AttentionHead_80178449482514
// MI455X (gfx1250) — compile-verified, matrix operand order checked by reading, not run on hardware
//
#include <hip/hip_runtime.h>
#include <hip/hip_bf16.h>

// ---------------- problem constants ----------------
constexpr int Bb = 2;
constexpr int Nn = 8192;
constexpr int Fn = 256;
constexpr int Dn = 128;
constexpr float ALPHA = 0.2f;

// ---------------- types ----------------
typedef __attribute__((ext_vector_type(16))) __bf16 v16bf;
typedef __attribute__((ext_vector_type(8)))  float  v8f;
typedef __attribute__((ext_vector_type(4)))  unsigned int u32x4;
typedef __attribute__((ext_vector_type(4)))  int i32x4;
typedef __attribute__((ext_vector_type(8)))  int i32x8;

struct alignas(16) U128 { unsigned x, y, z, w; };
struct alignas(16) F128 { float x, y, z, w; };
struct alignas(8)  U64  { unsigned x, y; };

union ABpack {
    v16bf v;
    U128 q[2];
    unsigned u[8];
    unsigned short s[16];
};
union US8 {
    unsigned short s[8];
    U128 q;
};

__device__ __forceinline__ unsigned short f2bf(float x) {
    unsigned u = __float_as_uint(x);
    u += 0x7fffu + ((u >> 16) & 1u);   // round-to-nearest-even
    return (unsigned short)(u >> 16);
}
__device__ __forceinline__ float bf2f(unsigned short s) {
    return __uint_as_float(((unsigned)s) << 16);
}
__device__ __forceinline__ float lrelu(float x) {
    return fmaxf(x, ALPHA * x);        // valid for 0 < ALPHA < 1
}

// ---------------- TDM support (CDNA5 Tensor Data Mover) ----------------
#if defined(__HIP_DEVICE_COMPILE__) && __has_builtin(__builtin_amdgcn_tensor_load_to_lds)
#define ATTN_USE_TDM 1
#endif

__device__ __forceinline__ void tdm_wait0() {
#if defined(__HIP_DEVICE_COMPILE__)
#if __has_builtin(__builtin_amdgcn_s_wait_tensorcnt)
    __builtin_amdgcn_s_wait_tensorcnt(0);
#else
    asm volatile("s_wait_tensorcnt 0x0" ::: "memory");
#endif
#endif
}

#if defined(ATTN_USE_TDM)
// Issue one TDM load of a 2D tile: 32 bf16 (j) x 128 rows (d), row stride N
// elements in global memory.  LDS pad: +16B after each 64B row => 80B LDS row
// stride (bank-conflict-free for the 16B ds_load_b128 operand fetches).
__device__ __forceinline__ void tdm_load_vtile(const unsigned short* gsrc, unsigned lds_off) {
    unsigned long long ga = (unsigned long long)(size_t)gsrc;
    u32x4 g0;
    g0[0] = 1u;                                   // count=1 valid descriptor
    g0[1] = lds_off;                              // lds_addr (bytes)
    g0[2] = (unsigned)ga;                         // global_addr[31:0]
    g0[3] = (unsigned)((ga >> 32) & 0x01ffffffu)  // global_addr[56:32]
          | 0x80000000u;                          // type=2 ("image") at [127:126]
    i32x8 g1;
    g1[0] = (1 << 16)                             // data_size = 2 bytes
          | (1 << 20)                             // pad_enable
          | (3 << 22)                             // pad_interval: every 64B
          | (3 << 25);                            // pad_amount: 16B
    g1[1] = (int)(((unsigned)Nn & 0xffffu) << 16);// tensor_dim0[15:0] (=8192)
    g1[2] = (int)((unsigned)Dn << 16);            // tensor_dim1[15:0] (=128)
    g1[3] = (int)(32u << 16);                     // tile_dim0 = 32
    g1[4] = Dn;                                   // tile_dim1 = 128 (tile_dim2=0)
    g1[5] = Nn;                                   // tensor_dim0_stride[31:0]
    g1[6] = 0;                                    // stride hi / dim1_stride lo
    g1[7] = 0;
    i32x4 gz = {0, 0, 0, 0};
#if __has_include(<hip/amd_detail/amd_gfx1250_TDM.h>)
    i32x8 gz8 = {0, 0, 0, 0, 0, 0, 0, 0};
    __builtin_amdgcn_tensor_load_to_lds(g0, g1, gz, gz, gz8, 0);
#else
    __builtin_amdgcn_tensor_load_to_lds(g0, g1, gz, gz, 0);
#endif
}
#endif

// Fallback cooperative copy (also used by host-side parse): 8KB tile, 128 thr.
__device__ __forceinline__ void copy_tile_sync(unsigned short (*dst)[40],
                                               const unsigned short* __restrict__ vbase,
                                               int j0, int tid) {
#pragma unroll
    for (int k = 0; k < 4; ++k) {
        int c = tid + 128 * k;           // 0..511 chunks of 16B
        int d = c >> 2, jc = (c & 3) * 8;
        *(U128*)&dst[d][jc] = *(const U128*)(vbase + (size_t)d * Nn + j0 + jc);
    }
}

// ============ kernel 1: seq f32 -> bf16 (flat, 4 elems/thread) ============
__global__ void cvt_seq_kernel(const float* __restrict__ in, unsigned short* __restrict__ out) {
    int t = blockIdx.x * blockDim.x + threadIdx.x;
    F128 v = *(const F128*)(in + 4 * (size_t)t);
    U64 o;
    o.x = (unsigned)f2bf(v.x) | ((unsigned)f2bf(v.y) << 16);
    o.y = (unsigned)f2bf(v.z) | ((unsigned)f2bf(v.w) << 16);
    *(U64*)(out + 4 * (size_t)t) = o;
}

// ============ kernel 2: Wf [F,D] f32 -> WfT [D,F] bf16 ============
__global__ void cvt_wf_kernel(const float* __restrict__ Wf, unsigned short* __restrict__ WfT) {
    int idx = blockIdx.x * blockDim.x + threadIdx.x;
    int d = idx / Fn, f = idx % Fn;
    WfT[idx] = f2bf(Wf[f * Dn + d]);
}

// ============ kernel 3: GEMM1  sfts = seq @ Wf  -> sftsT [B][D][N] bf16 ====
__global__ void gemm1_kernel(const unsigned short* __restrict__ seq_bf,   // [B*N, F]
                             const unsigned short* __restrict__ WfT_bf,   // [D, F]
                             unsigned short* __restrict__ sftsT_bf) {     // [B, D, N]
    const int wid  = threadIdx.x >> 5;
    const int lane = threadIdx.x & 31;
    const int ln   = lane & 15;
    const int hi   = lane >> 4;
    const int kb   = hi * 8;
    const int m0   = blockIdx.x * 16;
    const int d0   = wid * 16;

    const unsigned short* aptr = seq_bf + (size_t)(m0 + ln) * Fn;
    const unsigned short* bptr = WfT_bf + (size_t)(d0 + ln) * Fn;

    const v8f vzero = {0.f, 0.f, 0.f, 0.f, 0.f, 0.f, 0.f, 0.f};
    v8f acc = vzero;

#pragma unroll
    for (int k0 = 0; k0 < Fn; k0 += 32) {
        ABpack A, Bp;
        A.q[0]  = *(const U128*)(aptr + k0 + kb);
        A.q[1]  = *(const U128*)(aptr + k0 + kb + 16);
        Bp.q[0] = *(const U128*)(bptr + k0 + kb);
        Bp.q[1] = *(const U128*)(bptr + k0 + kb + 16);
        acc = __builtin_amdgcn_wmma_f32_16x16x32_bf16(false, A.v, false, Bp.v,
                                                      (short)0, acc, false, false);
    }

    const int b    = m0 / Nn;
    const int iloc = (m0 % Nn) + hi * 8;
    const int d    = d0 + ln;
    US8 o;
#pragma unroll
    for (int r = 0; r < 8; ++r) o.s[r] = f2bf(acc[r]);
    *(U128*)(sftsT_bf + ((size_t)(b * Dn + d)) * Nn + iloc) = o.q;
}

// ============ kernel 4: f1/f2 per-row scalars ============
__global__ void f12_kernel(const unsigned short* __restrict__ sftsT,
                           const float* __restrict__ w1, const float* __restrict__ b1,
                           const float* __restrict__ w2, const float* __restrict__ b2,
                           float* __restrict__ f1, float* __restrict__ f2) {
    int t = blockIdx.x * blockDim.x + threadIdx.x;
    int b = t / Nn, i = t % Nn;
    const unsigned short* p = sftsT + (size_t)b * Dn * Nn + i;
    float a1 = 0.f, a2 = 0.f;
#pragma unroll 4
    for (int d = 0; d < Dn; ++d) {
        float v = bf2f(p[(size_t)d * Nn]);
        a1 += v * w1[d];
        a2 += v * w2[d];
    }
    f1[t] = a1 + b1[0];
    f2[t] = a2 + b2[0];
}

// ============ kernel 5: per-batch max of f2 ============
__global__ void maxf2_kernel(const float* __restrict__ f2, float* __restrict__ M2) {
    __shared__ float red[256];
    const int b = blockIdx.x;
    float m = -3.402823466e38f;
    for (int j = threadIdx.x; j < Nn; j += 256) m = fmaxf(m, f2[b * Nn + j]);
    red[threadIdx.x] = m;
    __syncthreads();
    for (int s = 128; s > 0; s >>= 1) {
        if (threadIdx.x < s) red[threadIdx.x] = fmaxf(red[threadIdx.x], red[threadIdx.x + s]);
        __syncthreads();
    }
    if (threadIdx.x == 0) M2[b] = red[0];
}

// ============ kernel 6: fused softmax-attention GEMM ============
// Block = 128 threads (4 waves).  Wave: 16 rows (i) x D=128 cols, K loop over
// j.  V tiles (32j x 128d bf16, shared by all 4 waves) are double-buffered in
// LDS via TDM tensor_load_to_lds (wave 0 issues; s_wait_tensorcnt + barrier).
__global__ void attn_kernel(const unsigned short* __restrict__ sftsT,  // [B, D, N] bf16
                            const float* __restrict__ f1,
                            const float* __restrict__ f2,
                            const float* __restrict__ M2,
                            const float* __restrict__ bias,
                            float* __restrict__ out) {               // [B, N, D] f32
    const int wid  = threadIdx.x >> 5;
    const int lane = threadIdx.x & 31;
    const int ln   = lane & 15;
    const int hi   = lane >> 4;
    const int kb   = hi * 8;
    const int m0   = (blockIdx.x * 4 + wid) * 16;
    const int b    = m0 / Nn;
    const int i0   = m0 % Nn;

    // V tile double buffer: rows padded 32 -> 40 shorts (80B stride) so the
    // 16B ds_load_b128 operand reads are LDS bank-conflict free.
    __shared__ __align__(16) unsigned short vsh[2][Dn][40];
    __shared__ float s_sh[4 * 16];

    const float f1v  = f1[m0 + ln];
    const float mrow = lrelu(f1v + M2[b]);
    const float* f2b = f2 + (size_t)b * Nn;
    const unsigned short* vbase = sftsT + (size_t)b * Dn * Nn;

    const v8f vzero = {0.f, 0.f, 0.f, 0.f, 0.f, 0.f, 0.f, 0.f};
    v8f acc[8];
#pragma unroll
    for (int t = 0; t < 8; ++t) acc[t] = vzero;
    float lane_s = 0.f;

    // ---- prologue: fill buffer 0 with tile j0=0 ----
#if defined(ATTN_USE_TDM)
    if (wid == 0) {
        tdm_load_vtile(vbase, (unsigned)(size_t)&vsh[0][0][0]);
        tdm_wait0();
    }
#else
    copy_tile_sync(vsh[0], vbase, 0, threadIdx.x);
#endif
    __syncthreads();

    int cur = 0;
    for (int j0 = 0; j0 < Nn; j0 += 32) {
        const int nb = cur ^ 1;
        // ---- kick off async fill of the next tile ----
#if defined(ATTN_USE_TDM)
        if (wid == 0 && j0 + 32 < Nn)
            tdm_load_vtile(vbase + (j0 + 32), (unsigned)(size_t)&vsh[nb][0][0]);
#else
        if (j0 + 32 < Nn)
            copy_tile_sync(vsh[nb], vbase, j0 + 32, threadIdx.x);
#endif

        // ---- generate P tile (A operand, 16x32 bf16) in registers ----
        F128 qa = *(const F128*)(f2b + j0 + kb);
        F128 qb = *(const F128*)(f2b + j0 + kb + 4);
        F128 qc = *(const F128*)(f2b + j0 + 16 + kb);
        F128 qd = *(const F128*)(f2b + j0 + 16 + kb + 4);
        if (j0 + 32 < Nn) __builtin_prefetch(f2b + j0 + 32 + kb, 0, 0);

        float pv[16] = {qa.x, qa.y, qa.z, qa.w, qb.x, qb.y, qb.z, qb.w,
                        qc.x, qc.y, qc.z, qc.w, qd.x, qd.y, qd.z, qd.w};
#pragma unroll
        for (int t = 0; t < 16; ++t) {
            float e = __expf(lrelu(f1v + pv[t]) - mrow);
            pv[t] = e;
            lane_s += e;
        }
        ABpack A;
#pragma unroll
        for (int t = 0; t < 8; ++t)
            A.u[t] = (unsigned)f2bf(pv[2 * t]) | ((unsigned)f2bf(pv[2 * t + 1]) << 16);

        // ---- 8 WMMAs: B operands streamed from LDS (ds_load_b128) ----
        const unsigned short (*vcur)[40] = vsh[cur];
#pragma unroll
        for (int t = 0; t < 8; ++t) {
            const unsigned short* bp = &vcur[t * 16 + ln][kb];
            ABpack Bv;
            Bv.q[0] = *(const U128*)(bp);
            Bv.q[1] = *(const U128*)(bp + 16);
            acc[t] = __builtin_amdgcn_wmma_f32_16x16x32_bf16(false, A.v, false, Bv.v,
                                                             (short)0, acc[t], false, false);
        }

        // ---- close the double-buffer handoff ----
#if defined(ATTN_USE_TDM)
        if (wid == 0) tdm_wait0();
#endif
        __syncthreads();
        cur = nb;
    }

    // ---- reduce softmax denominators: lanes l and l+16 cover same row ----
    float s_full = lane_s + __shfl_xor(lane_s, 16, 32);
    if (lane < 16) s_sh[wid * 16 + lane] = s_full;
    __syncthreads();

    // ---- normalize, bias, leaky_relu, store f32 output ----
#pragma unroll
    for (int t = 0; t < 8; ++t) {
        const int d  = t * 16 + ln;
        const float bv = bias[d];
#pragma unroll
        for (int r = 0; r < 8; ++r) {
            const int m = r + hi * 8;
            const float s   = s_sh[wid * 16 + m];
            const float val = acc[t][r] / s + bv;
            out[((size_t)b * Nn + i0 + m) * Dn + d] = lrelu(val);
        }
    }
}

// ============================ launch ============================
extern "C" void kernel_launch(void* const* d_in, const int* in_sizes, int n_in,
                              void* d_out, int out_size, void* d_ws, size_t ws_size,
                              hipStream_t stream) {
    const float* seq  = (const float*)d_in[0];
    const float* Wf   = (const float*)d_in[1];
    const float* w1   = (const float*)d_in[2];
    const float* b1   = (const float*)d_in[3];
    const float* w2   = (const float*)d_in[4];
    const float* b2   = (const float*)d_in[5];
    const float* bias = (const float*)d_in[6];
    float* out = (float*)d_out;

    char* ws = (char*)d_ws;
    unsigned short* seq_bf = (unsigned short*)(ws);                    //  8,388,608 B
    unsigned short* WfT    = (unsigned short*)(ws + 8388608);          //     65,536 B
    unsigned short* sftsT  = (unsigned short*)(ws + 8454144);          //  4,194,304 B
    float* f1              = (float*)(ws + 12648448);                  //     65,536 B
    float* f2              = (float*)(ws + 12713984);                  //     65,536 B
    float* M2              = (float*)(ws + 12779520);                  //          8 B

    cvt_seq_kernel<<<(Bb * Nn * Fn / 4) / 256, 256, 0, stream>>>(seq, seq_bf);
    cvt_wf_kernel<<<(Dn * Fn) / 256, 256, 0, stream>>>(Wf, WfT);
    gemm1_kernel<<<(Bb * Nn) / 16, 256, 0, stream>>>(seq_bf, WfT, sftsT);
    f12_kernel<<<(Bb * Nn) / 256, 256, 0, stream>>>(sftsT, w1, b1, w2, b2, f1, f2);
    maxf2_kernel<<<Bb, 256, 0, stream>>>(f2, M2);
    attn_kernel<<<(Bb * Nn) / 64, 128, 0, stream>>>(sftsT, f1, f2, M2, bias, out);
}
